// AttentionNet_57260503990483
// MI455X (gfx1250) — compile-verified, matrix operand order checked by reading, not run on hardware
//
#include <hip/hip_runtime.h>
#include <hip/hip_bf16.h>
#include <math.h>

// Shapes from the reference
#define BB 32
#define LL 128
#define TT 64
#define PP 64
#define DD 128
#define EPSN 1e-5f

typedef __attribute__((ext_vector_type(16))) __bf16 v16bf;
typedef __attribute__((ext_vector_type(8)))  float  v8f;

__device__ __forceinline__ v8f wmma_bf16(v16bf a, v16bf b, v8f c) {
  // D = A(16x32 bf16) x B(32x16 bf16) + C(16x16 f32)
  return __builtin_amdgcn_wmma_f32_16x16x32_bf16(false, a, false, b, (short)0, c,
                                                 false, false);
}

// Build 4 N-tile x 2 K-step B fragments of the 64x64 matrix M, used as
// B[k][n] = M[n][k]  (Z = Y * M^T, i.e. z[p] = sum_q M[p][q] * y[q]).
// 16-bit fragment layout (lane L: col n=L&15, half h=L>>4):
//   element e holds K = 16*(e>>3) + 8*h + (e&7)  (+32 per K-step).
__device__ __forceinline__ void build_bfrags(const float* __restrict__ M, int lane,
                                             v16bf bf[4][2]) {
  const int n = lane & 15, h = lane >> 4;
#pragma unroll
  for (int nt = 0; nt < 4; ++nt)
#pragma unroll
    for (int kt = 0; kt < 2; ++kt) {
      v16bf f;
#pragma unroll
      for (int e = 0; e < 16; ++e) {
        const int k = kt * 32 + 16 * (e >> 3) + 8 * h + (e & 7);
        f[e] = (__bf16)M[(nt * 16 + n) * PP + k];
      }
      bf[nt][kt] = f;
    }
}

// ---------------------------------------------------------------------------
// Kernel 1: score vectors  sc/ts/es (B,L) via e_t@W@E^T, ec (B,L), tc (B,T)
// ---------------------------------------------------------------------------
__global__ void __launch_bounds__(256)
precompute_kernel(const float* __restrict__ e_t, const float* __restrict__ t_t,
                  const float* __restrict__ f_t, const float* __restrict__ E,
                  const float* __restrict__ Tv, const float* __restrict__ Fv,
                  const float* __restrict__ Ws1, const float* __restrict__ Wt1,
                  const float* __restrict__ We1, const float* __restrict__ We3,
                  const float* __restrict__ Wt3,
                  float* __restrict__ sc, float* __restrict__ ts,
                  float* __restrict__ es, float* __restrict__ ec,
                  float* __restrict__ tc) {
  const int b = blockIdx.x, tid = threadIdx.x;
  __shared__ float tS[256], tT[256], tE[256], tEc[128], tTc[128];
  {
    float s1 = 0.f, s2 = 0.f, s3 = 0.f;
    for (int d = 0; d < 2 * DD; ++d) {
      const float ev = e_t[b * 2 * DD + d];
      s1 += ev * Ws1[d * 2 * DD + tid];
      s2 += ev * Wt1[d * 2 * DD + tid];
      s3 += ev * We1[d * 2 * DD + tid];
    }
    tS[tid] = s1; tT[tid] = s2; tE[tid] = s3;
  }
  if (tid < DD) {
    float s4 = 0.f, s5 = 0.f;
    for (int d = 0; d < DD; ++d) {
      s4 += f_t[b * DD + d] * We3[d * DD + tid];
      s5 += t_t[b * DD + d] * Wt3[d * DD + tid];
    }
    tEc[tid] = s4; tTc[tid] = s5;
  }
  __syncthreads();
  if (tid < LL) {
    const int l = tid;
    float s1 = 0.f, s2 = 0.f, s3 = 0.f, s4 = 0.f;
    for (int e = 0; e < 2 * DD; ++e) {
      const float Ev = E[l * 2 * DD + e];
      s1 += tS[e] * Ev; s2 += tT[e] * Ev; s3 += tE[e] * Ev;
    }
    for (int e = 0; e < DD; ++e) s4 += tEc[e] * Fv[l * DD + e];
    sc[b * LL + l] = s1; ts[b * LL + l] = s2;
    es[b * LL + l] = s3; ec[b * LL + l] = s4;
  } else if (tid < LL + TT) {
    const int t = tid - LL;
    float s5 = 0.f;
    for (int e = 0; e < DD; ++e) s5 += tTc[e] * Tv[(b * TT + t) * DD + e];
    tc[b * TT + t] = s5;
  }
}

// ---------------------------------------------------------------------------
// Kernel 2: BN2d stats per channel l for the three branches, folded into
// affine n(v) = v*a_l + c_l.  One block per l.
// ---------------------------------------------------------------------------
__global__ void __launch_bounds__(256)
bnstats_kernel(const float* __restrict__ X, const float* __restrict__ sc,
               const float* __restrict__ ts, const float* __restrict__ es,
               const float* __restrict__ ec, const float* __restrict__ tc,
               const float* __restrict__ gS, const float* __restrict__ bS,
               const float* __restrict__ gT, const float* __restrict__ bT,
               const float* __restrict__ gE, const float* __restrict__ bE,
               float* aS, float* cS, float* aT, float* cT, float* aE, float* cE) {
  const int l = blockIdx.x, tid = threadIdx.x;
  __shared__ float red[256];
  auto reduce = [&](float v) -> float {
    red[tid] = v; __syncthreads();
    for (int st = 128; st > 0; st >>= 1) {
      if (tid < st) red[tid] += red[tid + st];
      __syncthreads();
    }
    const float r = red[0]; __syncthreads();
    return r;
  };
  float sx = 0.f, sx2 = 0.f;
  for (int i = tid; i < BB * TT; i += 256) {
    const int bi = i >> 6, t = i & 63;
    const float v = X[(bi * LL + l) * TT + t];
    sx += v; sx2 += v * v;
  }
  float stc = 0.f, stc2 = 0.f;
  for (int i = tid; i < BB * TT; i += 256) {
    const float v = tc[i]; stc += v; stc2 += v * v;
  }
  float ssc = 0.f, ssc2 = 0.f, sts = 0.f, sts2 = 0.f;
  float ses = 0.f, ses2 = 0.f, sec = 0.f, sec2 = 0.f;
  for (int bi = tid; bi < BB; bi += 256) {
    float v = sc[bi * LL + l]; ssc += v; ssc2 += v * v;
    v = ts[bi * LL + l]; sts += v; sts2 += v * v;
    v = es[bi * LL + l]; ses += v; ses2 += v * v;
    v = ec[bi * LL + l]; sec += v; sec2 += v * v;
  }
  sx = reduce(sx);  sx2 = reduce(sx2);
  stc = reduce(stc); stc2 = reduce(stc2);
  ssc = reduce(ssc); ssc2 = reduce(ssc2);
  sts = reduce(sts); sts2 = reduce(sts2);
  ses = reduce(ses); ses2 = reduce(ses2);
  sec = reduce(sec); sec2 = reduce(sec2);
  if (tid == 0) {
    {  // spatial: channels {sc (xT), X}; count = B*T*2
      const float n = (float)(BB * TT * 2);
      const float m = (TT * ssc + sx) / n;
      const float var = (TT * ssc2 + sx2) / n - m * m;
      const float a = gS[l] * rsqrtf(var + EPSN);
      aS[l] = a; cS[l] = bS[l] - m * a;
    }
    {  // temporal: channels {ts (xT), tc (over b,t), X}; count = B*T*3
      const float n = (float)(BB * TT * 3);
      const float m = (TT * sts + stc + sx) / n;
      const float var = (TT * sts2 + stc2 + sx2) / n - m * m;
      const float a = gT[l] * rsqrtf(var + EPSN);
      aT[l] = a; cT[l] = bT[l] - m * a;
    }
    {  // external: channels {es (xT), ec (xT), X}; count = B*T*3
      const float n = (float)(BB * TT * 3);
      const float m = (TT * ses + TT * sec + sx) / n;
      const float var = (TT * ses2 + TT * sec2 + sx2) / n - m * m;
      const float a = gE[l] * rsqrtf(var + EPSN);
      aE[l] = a; cE[l] = bE[l] - m * a;
    }
  }
}

// ---------------------------------------------------------------------------
// Kernel 3: fused branch with softmax over L (spatial C=2, external C=3).
// One block per b; 8 waves each own a 16-row (l) M-tile; per t:
//   BN->Linear->tanh built directly into bf16 A fragments, V-GEMM via WMMA,
//   softmax over l in LDS, h accumulated in registers.
// ---------------------------------------------------------------------------
__global__ void __launch_bounds__(256)
branchL_kernel(const float* __restrict__ u, const float* __restrict__ v,
               const float* __restrict__ X, const float* __restrict__ W,
               const float* __restrict__ bias, const float* __restrict__ Vmat,
               const float* __restrict__ bnA, const float* __restrict__ bnC,
               float* __restrict__ hout, int C) {
  const int b = blockIdx.x, tid = threadIdx.x;
  const int lane = tid & 31, wave = tid >> 5;
  const int m = lane & 15, h = lane >> 4;
  const int l_row = wave * 16 + m;

  __shared__ float z[LL][PP + 1];
  __shared__ float Wc[3][PP];
  __shared__ float biassh[PP];
  __shared__ float xcol[LL];
  __shared__ float red[4][PP];

  if (tid < PP) {
    for (int c = 0; c < C; ++c) Wc[c][tid] = W[tid * C + c];
    biassh[tid] = bias[tid];
  }
  v16bf bfrag[4][2];
  build_bfrags(Vmat, lane, bfrag);

  const float a_l = bnA[l_row], c_l = bnC[l_row];
  const float nu = u[b * LL + l_row] * a_l + c_l;
  const float nv = (C == 3) ? (v[b * LL + l_row] * a_l + c_l) : 0.f;

  const int p = tid & 63, g = tid >> 6;  // softmax role: column p, l-group g
  float hacc[32];
#pragma unroll
  for (int i = 0; i < 32; ++i) hacc[i] = 0.f;

  __syncthreads();

  for (int t = 0; t < TT; ++t) {
    if (tid < LL) {
      xcol[tid] = X[(b * LL + tid) * TT + t];
      if (t + 1 < TT) __builtin_prefetch(&X[(b * LL + tid) * TT + t + 1], 0, 0);
    }
    __syncthreads();
    const float nx = xcol[l_row] * a_l + c_l;
    v16bf afr[2];
#pragma unroll
    for (int kt = 0; kt < 2; ++kt) {
      v16bf f;
#pragma unroll
      for (int e = 0; e < 16; ++e) {
        const int q = kt * 32 + 16 * (e >> 3) + 8 * h + (e & 7);
        float y = nu * Wc[0][q] + nx * Wc[C - 1][q] + biassh[q];
        if (C == 3) y += nv * Wc[1][q];
        f[e] = (__bf16)tanhf(y);
      }
      afr[kt] = f;
    }
#pragma unroll
    for (int nt = 0; nt < 4; ++nt) {
      v8f d = {};
      d = wmma_bf16(afr[0], bfrag[nt][0], d);
      d = wmma_bf16(afr[1], bfrag[nt][1], d);
#pragma unroll
      for (int r = 0; r < 8; ++r)
        z[wave * 16 + r + 8 * h][nt * 16 + m] = d[r];
    }
    __syncthreads();
    // softmax over l (axis=1) per column p, 4 groups of 32 l's
    float mx = -1e30f;
    for (int i = 0; i < 32; ++i) mx = fmaxf(mx, z[g * 32 + i][p]);
    red[g][p] = mx;
    __syncthreads();
    mx = fmaxf(fmaxf(red[0][p], red[1][p]), fmaxf(red[2][p], red[3][p]));
    __syncthreads();
    float s = 0.f;
    for (int i = 0; i < 32; ++i) {
      const float ev = __expf(z[g * 32 + i][p] - mx);
      z[g * 32 + i][p] = ev;
      s += ev;
    }
    red[g][p] = s;
    __syncthreads();
    s = red[0][p] + red[1][p] + red[2][p] + red[3][p];
    const float inv = 1.f / s;
#pragma unroll
    for (int i = 0; i < 32; ++i)
      hacc[i] += z[g * 32 + i][p] * inv * xcol[g * 32 + i];
    __syncthreads();
  }
#pragma unroll
  for (int i = 0; i < 32; ++i)
    hout[(b * LL + g * 32 + i) * PP + p] = hacc[i];
}

// ---------------------------------------------------------------------------
// Kernel 4: fused temporal branch (softmax over T).  One block per (b, l-pair);
// 8 waves = 2 l's x 4 t-tiles of 16 rows.
// ---------------------------------------------------------------------------
__global__ void __launch_bounds__(256)
branchT_kernel(const float* __restrict__ tsv, const float* __restrict__ tcv,
               const float* __restrict__ X, const float* __restrict__ W,
               const float* __restrict__ bias, const float* __restrict__ Vmat,
               const float* __restrict__ bnA, const float* __restrict__ bnC,
               float* __restrict__ hout) {
  const int b = blockIdx.x >> 6;
  const int l0 = (blockIdx.x & 63) * 2;
  const int tid = threadIdx.x, lane = tid & 31, wave = tid >> 5;
  const int m = lane & 15, h = lane >> 4;
  const int l_loc = wave >> 2, mt = wave & 3;
  const int l = l0 + l_loc;
  const int t_row = mt * 16 + m;

  __shared__ float z[2][TT][PP + 1];
  __shared__ float Wc[3][PP];
  __shared__ float biassh[PP];
  __shared__ float xr[2][TT];

  if (tid < PP) {
    for (int c = 0; c < 3; ++c) Wc[c][tid] = W[tid * 3 + c];
    biassh[tid] = bias[tid];
  }
  if (tid < 2 * TT)
    xr[tid >> 6][tid & 63] = X[(b * LL + l0 + (tid >> 6)) * TT + (tid & 63)];

  v16bf bfrag[4][2];
  build_bfrags(Vmat, lane, bfrag);

  const float a_l = bnA[l], c_l = bnC[l];
  const float nts = tsv[b * LL + l] * a_l + c_l;
  const float ntc = tcv[b * TT + t_row] * a_l + c_l;
  __syncthreads();
  const float nx = xr[l_loc][t_row] * a_l + c_l;

  v16bf afr[2];
#pragma unroll
  for (int kt = 0; kt < 2; ++kt) {
    v16bf f;
#pragma unroll
    for (int e = 0; e < 16; ++e) {
      const int q = kt * 32 + 16 * (e >> 3) + 8 * h + (e & 7);
      const float y = nts * Wc[0][q] + ntc * Wc[1][q] + nx * Wc[2][q] + biassh[q];
      f[e] = (__bf16)tanhf(y);
    }
    afr[kt] = f;
  }
#pragma unroll
  for (int nt = 0; nt < 4; ++nt) {
    v8f d = {};
    d = wmma_bf16(afr[0], bfrag[nt][0], d);
    d = wmma_bf16(afr[1], bfrag[nt][1], d);
#pragma unroll
    for (int r = 0; r < 8; ++r)
      z[l_loc][mt * 16 + r + 8 * h][nt * 16 + m] = d[r];
  }
  __syncthreads();
  if (tid < 128) {  // softmax over t (axis=2) + weighted sum; column (l_loc,p)
    const int ll = tid >> 6, p = tid & 63;
    float mx = -1e30f;
    for (int t = 0; t < TT; ++t) mx = fmaxf(mx, z[ll][t][p]);
    float s = 0.f;
    for (int t = 0; t < TT; ++t) {
      const float ev = __expf(z[ll][t][p] - mx);
      z[ll][t][p] = ev;
      s += ev;
    }
    const float inv = 1.f / s;
    float o = 0.f;
    for (int t = 0; t < TT; ++t) o += z[ll][t][p] * inv * xr[ll][t];
    hout[(b * LL + l0 + ll) * PP + p] = o;
  }
}

// ---------------------------------------------------------------------------
// Kernel 5: fusion  h=(hs,ht,he) -> BN2d(ch=l) -> 3->1 linear -> ReLU
// ---------------------------------------------------------------------------
__global__ void __launch_bounds__(256)
fusion_kernel(const float* __restrict__ hs, const float* __restrict__ ht,
              const float* __restrict__ he, const float* __restrict__ g,
              const float* __restrict__ bb, const float* __restrict__ Wf,
              const float* __restrict__ bf, float* __restrict__ hout) {
  const int l = blockIdx.x, tid = threadIdx.x;
  __shared__ float red[256];
  float s = 0.f, s2 = 0.f;
  for (int i = tid; i < BB * PP; i += 256) {
    const int bi = i >> 6, p = i & 63;
    const int idx = (bi * LL + l) * PP + p;
    const float a1 = hs[idx], a2 = ht[idx], a3 = he[idx];
    s += a1 + a2 + a3;
    s2 += a1 * a1 + a2 * a2 + a3 * a3;
  }
  red[tid] = s; __syncthreads();
  for (int st = 128; st > 0; st >>= 1) {
    if (tid < st) red[tid] += red[tid + st];
    __syncthreads();
  }
  s = red[0]; __syncthreads();
  red[tid] = s2; __syncthreads();
  for (int st = 128; st > 0; st >>= 1) {
    if (tid < st) red[tid] += red[tid + st];
    __syncthreads();
  }
  s2 = red[0]; __syncthreads();
  const float n = (float)(BB * PP * 3);
  const float mean = s / n, var = s2 / n - mean * mean;
  const float a = g[l] * rsqrtf(var + EPSN), c = bb[l] - mean * a;
  const float w0 = Wf[0], w1 = Wf[1], w2 = Wf[2], b0 = bf[0];
  for (int i = tid; i < BB * PP; i += 256) {
    const int bi = i >> 6, p = i & 63;
    const int idx = (bi * LL + l) * PP + p;
    const float o = (hs[idx] * a + c) * w0 + (ht[idx] * a + c) * w1 +
                    (he[idx] * a + c) * w2 + b0;
    hout[idx] = fmaxf(o, 0.f);
  }
}

// ---------------------------------------------------------------------------
// FC head: BN1d stats (over batch of 32) folded into affine, then GEMV+ReLU.
// ---------------------------------------------------------------------------
__global__ void bn1dstats_kernel(const float* __restrict__ h,
                                 const float* __restrict__ tinfo,
                                 const float* __restrict__ g,
                                 const float* __restrict__ bb,
                                 float* __restrict__ aF, float* __restrict__ cF) {
  const int F = LL * PP + 4;
  const int f = blockIdx.x * 256 + threadIdx.x;
  if (f >= F) return;
  float s = 0.f, s2 = 0.f;
  for (int b = 0; b < BB; ++b) {
    const float v = (f < LL * PP) ? h[b * LL * PP + f] : tinfo[b * 4 + (f - LL * PP)];
    s += v; s2 += v * v;
  }
  const float m = s / BB, var = s2 / BB - m * m;
  const float a = g[f] * rsqrtf(var + EPSN);
  aF[f] = a; cF[f] = bb[f] - m * a;
}

__global__ void __launch_bounds__(256)
fc1_kernel(const float* __restrict__ h, const float* __restrict__ tinfo,
           const float* __restrict__ aF, const float* __restrict__ cF,
           const float* __restrict__ W, const float* __restrict__ bias,
           float* __restrict__ xout) {
  const int b = blockIdx.x, tid = threadIdx.x;
  const int F = LL * PP + 4;  // 8196 = 4 * 2049
  __shared__ float cat[LL * PP + 4];
  __shared__ float red[4][PP];
  for (int f = tid; f < F; f += 256) {
    const float v = (f < LL * PP) ? h[b * LL * PP + f] : tinfo[b * 4 + (f - LL * PP)];
    cat[f] = v * aF[f] + cF[f];
  }
  __syncthreads();
  const int p = tid & 63, q = tid >> 6;
  float acc = 0.f;
  for (int f = q * 2049; f < (q + 1) * 2049; ++f) acc += cat[f] * W[p * F + f];
  red[q][p] = acc;
  __syncthreads();
  if (q == 0) {
    const float tot = red[0][p] + red[1][p] + red[2][p] + red[3][p] + bias[p];
    xout[b * PP + p] = fmaxf(tot, 0.f);
  }
}

__global__ void fc2_kernel(const float* __restrict__ x1, const float* __restrict__ x2,
                           const float* __restrict__ g, const float* __restrict__ bb,
                           const float* __restrict__ W, const float* __restrict__ bias,
                           float* __restrict__ out) {
  const int tid = threadIdx.x;  // 128 threads
  __shared__ float aF[128], cF[128];
  {
    float s = 0.f, s2 = 0.f;
    for (int b = 0; b < BB; ++b) {
      const float v = (tid < 64) ? x1[b * 64 + tid] : x2[b * 64 + tid - 64];
      s += v; s2 += v * v;
    }
    const float m = s / BB, var = s2 / BB - m * m;
    const float a = g[tid] * rsqrtf(var + EPSN);
    aF[tid] = a; cF[tid] = bb[tid] - m * a;
  }
  __syncthreads();
  if (tid < BB) {
    float acc = bias[0];
    for (int f = 0; f < 128; ++f) {
      const float v = (f < 64) ? x1[tid * 64 + f] : x2[tid * 64 + f - 64];
      acc += (v * aF[f] + cF[f]) * W[f];
    }
    out[tid] = fmaxf(acc, 0.f);
  }
}

// ---------------------------------------------------------------------------
// Launch. params flattened jax-pytree style (alphabetical keys at each level):
//  top: fc1_1, fc1_2, fc2_1, layer1, layer2; lin: W, b, bn_b, bn_g;
//  layer: Vs, Vt, We1, We3, Ws1, Wt1, Wt3, external(4), fusion(4),
//         spatial(4), temporal(4).
// ---------------------------------------------------------------------------
extern "C" void kernel_launch(void* const* d_in, const int* in_sizes, int n_in,
                              void* d_out, int out_size, void* d_ws, size_t ws_size,
                              hipStream_t stream) {
  (void)in_sizes; (void)n_in; (void)out_size; (void)ws_size;
  const float* Xin[2] = {(const float*)d_in[0], (const float*)d_in[1]};
  const float* e_t = (const float*)d_in[2];
  const float* t_t = (const float*)d_in[3];
  const float* f_t = (const float*)d_in[4];
  const float* E = (const float*)d_in[5];
  const float* Tv = (const float*)d_in[6];
  const float* Fv = (const float*)d_in[7];
  const float* tinfo = (const float*)d_in[8];
  const int FC11 = 9, FC12 = 13, FC21 = 17;
  const int LYB[2] = {21, 44};
  auto P_ = [&](int i) { return (const float*)d_in[i]; };

  float* ws = (float*)d_ws;
  float* sc = ws;               // B*L
  float* tsb = ws + 4096;
  float* esb = ws + 8192;
  float* ecb = ws + 12288;
  float* tcb = ws + 16384;      // B*T
  float* aS = ws + 18432; float* cS = aS + LL;
  float* aT = cS + LL;    float* cT = aT + LL;
  float* aE = cT + LL;    float* cE = aE + LL;
  float* hs = ws + 19200;            // B*L*P
  float* ht = hs + BB * LL * PP;
  float* he = ht + BB * LL * PP;
  float* h12[2] = {he + BB * LL * PP, he + 2 * BB * LL * PP};
  float* aF1 = h12[1] + BB * LL * PP;
  float* cF1 = aF1 + (LL * PP + 4);
  float* aF2 = cF1 + (LL * PP + 4);
  float* cF2 = aF2 + (LL * PP + 4);
  float* x1 = cF2 + (LL * PP + 4);
  float* x2 = x1 + BB * PP;

  for (int ly = 0; ly < 2; ++ly) {
    const int base = LYB[ly];
    const float *Vs = P_(base + 0), *Vt = P_(base + 1);
    const float *We1 = P_(base + 2), *We3 = P_(base + 3);
    const float *Ws1 = P_(base + 4), *Wt1 = P_(base + 5), *Wt3 = P_(base + 6);
    const float *exW = P_(base + 7), *exB = P_(base + 8),
                *exBb = P_(base + 9), *exBg = P_(base + 10);
    const float *fuW = P_(base + 11), *fuB = P_(base + 12),
                *fuBb = P_(base + 13), *fuBg = P_(base + 14);
    const float *spW = P_(base + 15), *spB = P_(base + 16),
                *spBb = P_(base + 17), *spBg = P_(base + 18);
    const float *tmW = P_(base + 19), *tmB = P_(base + 20),
                *tmBb = P_(base + 21), *tmBg = P_(base + 22);

    precompute_kernel<<<BB, 256, 0, stream>>>(e_t, t_t, f_t, E, Tv, Fv, Ws1, Wt1,
                                              We1, We3, Wt3, sc, tsb, esb, ecb, tcb);
    bnstats_kernel<<<LL, 256, 0, stream>>>(Xin[ly], sc, tsb, esb, ecb, tcb,
                                           spBg, spBb, tmBg, tmBb, exBg, exBb,
                                           aS, cS, aT, cT, aE, cE);
    branchL_kernel<<<BB, 256, 0, stream>>>(sc, nullptr, Xin[ly], spW, spB, Vs,
                                           aS, cS, hs, 2);
    branchT_kernel<<<BB * LL / 2, 256, 0, stream>>>(tsb, tcb, Xin[ly], tmW, tmB,
                                                    Vt, aT, cT, ht);
    // NOTE: reference faithfully uses Vt (not Ve) for the external branch.
    branchL_kernel<<<BB, 256, 0, stream>>>(esb, ecb, Xin[ly], exW, exB, Vt,
                                           aE, cE, he, 3);
    fusion_kernel<<<LL, 256, 0, stream>>>(hs, ht, he, fuBg, fuBb, fuW, fuB, h12[ly]);
  }
  const int F = LL * PP + 4;
  bn1dstats_kernel<<<(F + 255) / 256, 256, 0, stream>>>(h12[0], tinfo, P_(FC11 + 3),
                                                        P_(FC11 + 2), aF1, cF1);
  bn1dstats_kernel<<<(F + 255) / 256, 256, 0, stream>>>(h12[1], tinfo, P_(FC12 + 3),
                                                        P_(FC12 + 2), aF2, cF2);
  fc1_kernel<<<BB, 256, 0, stream>>>(h12[0], tinfo, aF1, cF1, P_(FC11 + 0),
                                     P_(FC11 + 1), x1);
  fc1_kernel<<<BB, 256, 0, stream>>>(h12[1], tinfo, aF2, cF2, P_(FC12 + 0),
                                     P_(FC12 + 1), x2);
  fc2_kernel<<<1, 128, 0, stream>>>(x1, x2, P_(FC21 + 3), P_(FC21 + 2),
                                    P_(FC21 + 0), P_(FC21 + 1), (float*)d_out);
}
